// DGI_30855045054725
// MI455X (gfx1250) — matrix operand order repaired and checked by reading, not compiled, not run on hardware
//
#include <hip/hip_runtime.h>
#include <math.h>

#define NN 16384

typedef __bf16 v16bf __attribute__((ext_vector_type(16)));
typedef float  v8f   __attribute__((ext_vector_type(8)));
typedef unsigned int u32x4 __attribute__((ext_vector_type(4)));
typedef float  f32x4 __attribute__((ext_vector_type(4)));

union Frag16 { u32x4 q[2]; v16bf v; };
union Pack8  { __bf16 h[8]; u32x4 q; };

// ---------------------------------------------------------------------------
// adj fp32 -> bf16 (streamed once; 1 GB read + 0.5 GB write)
// ---------------------------------------------------------------------------
__global__ void conv_adj(const float* __restrict__ a, __bf16* __restrict__ b) {
    size_t i = ((size_t)blockIdx.x * 256 + threadIdx.x) * 8;
    f32x4 x = *reinterpret_cast<const f32x4*>(a + i);
    f32x4 y = *reinterpret_cast<const f32x4*>(a + i + 4);
    Pack8 o;
    o.h[0] = (__bf16)x.x; o.h[1] = (__bf16)x.y; o.h[2] = (__bf16)x.z; o.h[3] = (__bf16)x.w;
    o.h[4] = (__bf16)y.x; o.h[5] = (__bf16)y.y; o.h[6] = (__bf16)y.z; o.h[7] = (__bf16)y.w;
    *reinterpret_cast<u32x4*>(b + i) = o.q;
}

// ---------------------------------------------------------------------------
// Z^T[col][node] = sum_k H[node][k] * W[k][col]   (bf16, transposed so WMMA
// B-fragments are contiguous per lane). Padded cols (col>=dout) -> 0.
// grid: (Dp/16, NN/128, 2 sequences), block: 128
// ---------------------------------------------------------------------------
__global__ void feat_matmul(const float* __restrict__ H0, const float* __restrict__ H1,
                            int ldH, const float* __restrict__ W,
                            int din, int dout, int Dp, __bf16* __restrict__ Zt) {
    __shared__ float Ws[264 * 16];
    const int tid  = threadIdx.x;
    const int d0   = blockIdx.x * 16;
    const int node = blockIdx.y * 128 + tid;
    const float* H = blockIdx.z ? H1 : H0;
    __bf16* Zts    = Zt + (size_t)blockIdx.z * Dp * NN;

    for (int idx = tid; idx < din * 16; idx += 128) {
        int k = idx >> 4, j = idx & 15;
        int d = d0 + j;
        Ws[idx] = (d < dout) ? W[(size_t)k * dout + d] : 0.0f;
    }
    __syncthreads();

    const float* hrow = H + (size_t)node * ldH;
    float acc[16] = {};
    for (int k = 0; k < din; k++) {
        float hv = hrow[k];
        #pragma unroll
        for (int j = 0; j < 16; j++) acc[j] += hv * Ws[(k << 4) + j];
    }
    #pragma unroll
    for (int j = 0; j < 16; j++)
        Zts[(size_t)(d0 + j) * NN + node] = (__bf16)acc[j];
}

// ---------------------------------------------------------------------------
// H_out = relu(adj_bf16 @ Z + bias), both sequences fused as extra columns.
// Block: 512 threads = 16 waves = 2 row-groups x 8 column-groups; 64 rows per
// block (each wave: 2 m-subtiles sharing each B fragment).  The two row-
// groups stream identical B lines close together in time, so the second
// group's B loads hit the WGP cache (per-k-step B set = 40KB < 64KB WGP$) —
// B L2 traffic scales with M/64 instead of M/32.
// Column-tile count T = 8*NT (compile time): column-group wn owns tiles
// [wn*NT, wn*NT+NT) -> branch-free inner loop, explicit 2-stage software
// pipeline, peeled tail.  __launch_bounds__(512) => >=4 waves/SIMD => VGPRs
// capped at 256 (true live set ~210), avoiding MSB/live-range-split copies.
// A frag (16x32 bf16, ISA layout): lane = row; lo half K[0..8)+[16..24),
// hi half K[8..16)+[24..32). B frag from Z^T: lane = col, same K layout as A.
// ---------------------------------------------------------------------------
template<int NT>
struct Stage { Frag16 a0, a1, b[NT]; };

template<int NT>
__device__ __forceinline__ void load_stage(Stage<NT>& s,
                                           const __bf16* __restrict__ arow0,
                                           const __bf16* __restrict__ arow1,
                                           const __bf16* const (&bptr)[NT], int k0) {
    s.a0.q[0] = *reinterpret_cast<const u32x4*>(arow0 + k0);
    s.a0.q[1] = *reinterpret_cast<const u32x4*>(arow0 + k0 + 16);
    s.a1.q[0] = *reinterpret_cast<const u32x4*>(arow1 + k0);
    s.a1.q[1] = *reinterpret_cast<const u32x4*>(arow1 + k0 + 16);
    #pragma unroll
    for (int i = 0; i < NT; i++) {
        s.b[i].q[0] = *reinterpret_cast<const u32x4*>(bptr[i] + k0);
        s.b[i].q[1] = *reinterpret_cast<const u32x4*>(bptr[i] + k0 + 16);
    }
}

template<int NT>
__device__ __forceinline__ void compute_stage(Stage<NT>& s, v8f (&acc)[2][NT]) {
    #pragma unroll
    for (int i = 0; i < NT; i++) {
        acc[0][i] = __builtin_amdgcn_wmma_f32_16x16x32_bf16(
            false, s.a0.v, false, s.b[i].v, (short)0, acc[0][i], false, false);
        acc[1][i] = __builtin_amdgcn_wmma_f32_16x16x32_bf16(
            false, s.a1.v, false, s.b[i].v, (short)0, acc[1][i], false, false);
    }
}

template<int NT>
__global__ __launch_bounds__(512)
void gemm_adj(const __bf16* __restrict__ adjb, const __bf16* __restrict__ Zt,
              const float* __restrict__ bias, float* __restrict__ Hout,
              int Dp, int dout) {
    const int lane = threadIdx.x & 31;
    const int wave = threadIdx.x >> 5;     // 0..15
    const int wn   = wave & 7;             // column group
    const int wm   = wave >> 3;            // row group (0/1)
    const int hs   = lane >> 4;            // K-half select
    const int l15  = lane & 15;
    const int m0   = blockIdx.x * 64 + wm * 32;

    const __bf16* arow0 = adjb + (size_t)(m0 + l15) * NN + hs * 8;
    const __bf16* arow1 = arow0 + (size_t)16 * NN;

    const __bf16* bptr[NT];
    #pragma unroll
    for (int i = 0; i < NT; i++)
        bptr[i] = Zt + (size_t)((wn * NT + i) * 16 + l15) * NN + hs * 8;

    v8f acc[2][NT];
    #pragma unroll
    for (int i = 0; i < NT; i++) { acc[0][i] = {}; acc[1][i] = {}; }

    Stage<NT> sA, sB;
    load_stage<NT>(sA, arow0, arow1, bptr, 0);
    for (int k0 = 0; k0 < NN - 64; k0 += 64) {
        load_stage<NT>(sB, arow0, arow1, bptr, k0 + 32);
        compute_stage<NT>(sA, acc);
        load_stage<NT>(sA, arow0, arow1, bptr, k0 + 64);
        compute_stage<NT>(sB, acc);
    }
    // peeled final iteration (no wrap, no extra loads)
    load_stage<NT>(sB, arow0, arow1, bptr, NN - 32);
    compute_stage<NT>(sA, acc);
    compute_stage<NT>(sB, acc);

    // Epilogue: C/D layout — VGPR r, lanes0-15: M=r,N=lane; lanes16-31: M=8+r
    #pragma unroll
    for (int i = 0; i < NT; i++) {
        int col  = (wn * NT + i) * 16 + l15;
        int seq  = (col >= Dp) ? 1 : 0;
        int feat = col - seq * Dp;
        float bv = (feat < dout) ? bias[feat] : 0.0f;
        float* op = Hout + (size_t)seq * NN * Dp + feat;
        #pragma unroll
        for (int ms = 0; ms < 2; ms++) {
            #pragma unroll
            for (int r = 0; r < 8; r++) {
                int node = m0 + ms * 16 + hs * 8 + r;
                float v = acc[ms][i][r] + bv;
                op[(size_t)node * Dp] = v > 0.0f ? v : 0.0f;
            }
        }
    }
}

// ---------------------------------------------------------------------------
// c = sigmoid(mean over nodes of h1);  grid=64 (one block per feature)
// ---------------------------------------------------------------------------
__global__ void readout(const float* __restrict__ h1, float* __restrict__ c) {
    __shared__ float red[256];
    int j = blockIdx.x;
    float s = 0.0f;
    for (int n = threadIdx.x; n < NN; n += 256) s += h1[(size_t)n * 64 + j];
    red[threadIdx.x] = s;
    __syncthreads();
    for (int off = 128; off > 0; off >>= 1) {
        if ((int)threadIdx.x < off) red[threadIdx.x] += red[threadIdx.x + off];
        __syncthreads();
    }
    if (threadIdx.x == 0) {
        float m = red[0] * (1.0f / (float)NN);
        c[j] = 1.0f / (1.0f + expf(-m));
    }
}

// cw[e] = sum_d c[d] * wd[e][d]
__global__ void cw_kernel(const float* __restrict__ c, const float* __restrict__ wd,
                          float* __restrict__ cw) {
    int e = threadIdx.x;
    float s = 0.0f;
    for (int d = 0; d < 64; d++) s += c[d] * wd[e * 64 + d];
    cw[e] = s;
}

// sc[i] = h[seq][node] . cw + bd + samp_bias
__global__ void scores(const float* __restrict__ hf, const float* __restrict__ cw,
                       const float* __restrict__ sb1, const float* __restrict__ sb2,
                       const float* __restrict__ bd, float* __restrict__ out) {
    int i = blockIdx.x * 256 + threadIdx.x;       // i < 2*NN
    int seq = i >> 14;
    int node = i & (NN - 1);
    const float* hr = hf + ((size_t)seq * NN + node) * 64;
    float s = bd[0] + (seq ? sb2[node] : sb1[node]);
    #pragma unroll
    for (int d = 0; d < 64; d += 4) {
        f32x4 hv = *reinterpret_cast<const f32x4*>(hr + d);
        f32x4 cv = *reinterpret_cast<const f32x4*>(cw + d);
        s += hv.x * cv.x + hv.y * cv.y + hv.z * cv.z + hv.w * cv.w;
    }
    out[i] = s;
}

// ---------------------------------------------------------------------------
extern "C" void kernel_launch(void* const* d_in, const int* in_sizes, int n_in,
                              void* d_out, int out_size, void* d_ws, size_t ws_size,
                              hipStream_t stream) {
    const float* seq1 = (const float*)d_in[0];
    const float* seq2 = (const float*)d_in[1];
    const float* adj  = (const float*)d_in[2];
    const float* sb1  = (const float*)d_in[3];
    const float* sb2  = (const float*)d_in[4];
    const float* w1   = (const float*)d_in[5];
    const float* b1   = (const float*)d_in[6];
    const float* w2   = (const float*)d_in[7];
    const float* b2   = (const float*)d_in[8];
    const float* w3   = (const float*)d_in[9];
    const float* b3   = (const float*)d_in[10];
    const float* wd   = (const float*)d_in[11];
    const float* bd   = (const float*)d_in[12];
    float* out = (float*)d_out;

    // workspace carve-up (Dp padded to multiples of 64: 320 / 192 / 64)
    char* ws = (char*)d_ws;
    const size_t ADJB_SZ = (size_t)NN * NN * 2;            // 512 MB bf16 adj
    const size_t ZT_SZ   = (size_t)2 * 320 * NN * 2;       // Z^T (both seqs)
    const size_t H_SZ    = (size_t)2 * NN * 320 * 4;       // fp32 activations
    __bf16* adjb = (__bf16*)ws;
    __bf16* zt   = (__bf16*)(ws + ADJB_SZ);
    float*  ha   = (float*)(ws + ADJB_SZ + ZT_SZ);
    float*  hb   = (float*)(ws + ADJB_SZ + ZT_SZ + H_SZ);
    float*  cbuf = (float*)(ws + ADJB_SZ + ZT_SZ + 2 * H_SZ);
    float*  cwb  = cbuf + 64;

    // 1) adj -> bf16
    conv_adj<<<(unsigned)((size_t)NN * NN / 2048), 256, 0, stream>>>(adj, adjb);

    // layer 1: 64 -> 264 (Dp=320, T=40, NT=5)
    feat_matmul<<<dim3(320 / 16, NN / 128, 2), 128, 0, stream>>>(
        seq1, seq2, 64, w1, 64, 264, 320, zt);
    gemm_adj<5><<<NN / 64, 512, 0, stream>>>(adjb, zt, b1, ha, 320, 264);

    // layer 2: 264 -> 164 (Dp=192, T=24, NT=3)
    feat_matmul<<<dim3(192 / 16, NN / 128, 2), 128, 0, stream>>>(
        ha, ha + (size_t)NN * 320, 320, w2, 264, 164, 192, zt);
    gemm_adj<3><<<NN / 64, 512, 0, stream>>>(adjb, zt, b2, hb, 192, 164);

    // layer 3: 164 -> 64 (Dp=64, T=8, NT=1)
    feat_matmul<<<dim3(64 / 16, NN / 128, 2), 128, 0, stream>>>(
        hb, hb + (size_t)NN * 192, 192, w3, 164, 64, 64, zt);
    gemm_adj<1><<<NN / 64, 512, 0, stream>>>(adjb, zt, b3, ha, 64, 64);

    // readout + discriminator
    readout<<<64, 256, 0, stream>>>(ha, cbuf);               // from h_1 (seq 0)
    cw_kernel<<<1, 64, 0, stream>>>(cbuf, wd, cwb);
    scores<<<2 * NN / 256, 256, 0, stream>>>(ha, cwb, sb1, sb2, bd, out);
}
